// VectorQuantizer_27908697489601
// MI455X (gfx1250) — matrix operand order repaired and checked by reading, not compiled, not run on hardware
//
#include <hip/hip_runtime.h>

#define NUM_CODES 512
#define CODE_DIM  64
#define CB_STRIDE 72   // padded LDS row stride in f16 units (36 dwords) -> bank-conflict-free B loads

typedef __attribute__((ext_vector_type(16))) _Float16 v16h;
typedef __attribute__((ext_vector_type(8)))  float    v8f;

// One wave handles 16 rows x all 512 codes. 8 waves per block => 128 rows/block.
__global__ __launch_bounds__(256) void vq_wmma_kernel(
    const float* __restrict__ latents,    // [rows][64]
    const float* __restrict__ codebook,   // [512][64]
    float* __restrict__ out_q,            // [rows][64]
    float* __restrict__ out_codes,        // [rows] (written as float values)
    float* __restrict__ block_partials)   // [gridDim.x] SSE partials
{
    __shared__ _Float16 cb_hi[NUM_CODES * CB_STRIDE];  // 72 KB
    __shared__ _Float16 cb_lo[NUM_CODES * CB_STRIDE];  // 72 KB
    __shared__ float    c2s[NUM_CODES];                // 2 KB
    __shared__ int      wave_codes[8][16];
    __shared__ float    wave_partial[8];

    const int tid = threadIdx.x;

    // ---- Stage 1: codebook -> LDS as f16 hi/lo split (exact 2-term decomposition),
    //      rows padded to CB_STRIDE for conflict-free ds_load_b128 in the hot loop.
    const float4* cb4 = (const float4*)codebook;
    for (int i = tid; i < NUM_CODES * (CODE_DIM / 4); i += 256) {
        const int code = i >> 4;
        const int col4 = (i & 15) * 4;
        float4 v = cb4[i];
        _Float16* hi = &cb_hi[code * CB_STRIDE + col4];
        _Float16* lo = &cb_lo[code * CB_STRIDE + col4];
        _Float16 h0 = (_Float16)v.x, h1 = (_Float16)v.y,
                 h2 = (_Float16)v.z, h3 = (_Float16)v.w;
        hi[0] = h0; lo[0] = (_Float16)(v.x - (float)h0);
        hi[1] = h1; lo[1] = (_Float16)(v.y - (float)h1);
        hi[2] = h2; lo[2] = (_Float16)(v.z - (float)h2);
        hi[3] = h3; lo[3] = (_Float16)(v.w - (float)h3);
    }
    // ||c||^2 in fp32, vectorized & deterministic (L2-resident reload)
    for (int k = tid; k < NUM_CODES; k += 256) {
        const float4* r4 = (const float4*)(codebook + k * CODE_DIM);
        float s = 0.f;
        #pragma unroll
        for (int j = 0; j < CODE_DIM / 4; ++j) {
            float4 v = r4[j];
            s = fmaf(v.x, v.x, s); s = fmaf(v.y, v.y, s);
            s = fmaf(v.z, v.z, s); s = fmaf(v.w, v.w, s);
        }
        c2s[k] = s;
    }
    __syncthreads();

    // ---- Stage 2: per-wave A-matrix (16 rows), f16 hi/lo split in WMMA layout.
    const int wave = tid >> 5;
    const int lane = tid & 31;
    const int m0   = (blockIdx.x * 8 + wave) * 16;      // first row of this wave's tile
    const int m    = m0 + (lane & 15);                  // this lane's row (M)
    const int kb   = (lane >> 4) * 16;                  // K sub-offset from lane half

    const float* arow = latents + (size_t)m * CODE_DIM;
    v16h a_hi[2], a_lo[2];                              // two K-steps of 32 (D=64)
    #pragma unroll
    for (int s = 0; s < 2; ++s) {
        #pragma unroll
        for (int h = 0; h < 2; ++h) {                   // element 8h+j <-> K = kb+8h+j
            const float* p = arow + 32 * s + 8 * h + kb;
            #pragma unroll
            for (int j = 0; j < 8; ++j) {
                float x = p[j];
                _Float16 hi = (_Float16)x;
                a_hi[s][8*h + j] = hi;
                a_lo[s][8*h + j] = (_Float16)(x - (float)hi);
            }
        }
    }

    // ---- Stage 3: GEMM tiles + running argmin of (c2 - 2*x.c).
    //      Two independent accumulators per tile break the WMMA RAW chain;
    //      unroll-by-2 lets tile t+1's loads/WMMAs hide tile t's hazard window.
    const int kb2 = (lane >> 4) * 16;                   // B: element e <-> K = kb2 + e
    float minv[8];
    int   mini[8];
    #pragma unroll
    for (int r = 0; r < 8; ++r) { minv[r] = 3.402823466e38f; mini[r] = 0; }

    int boff = (lane & 15) * CB_STRIDE + kb2;           // strength-reduced B offset
    int n    = (lane & 15);                             // this lane's code column (N)
    #pragma unroll 2
    for (int t = 0; t < NUM_CODES / 16; ++t, boff += 16 * CB_STRIDE, n += 16) {
        const _Float16* bh0p = &cb_hi[boff];
        const _Float16* bl0p = &cb_lo[boff];
        const v16h bhi0 = *(const v16h*)(bh0p);
        const v16h bhi1 = *(const v16h*)(bh0p + 32);
        const v16h blo0 = *(const v16h*)(bl0p);
        const v16h blo1 = *(const v16h*)(bl0p + 32);

        v8f acc0 = {}, acc1 = {};
        acc0 = __builtin_amdgcn_wmma_f32_16x16x32_f16(false, a_hi[0], false, bhi0,
                                                      (short)0, acc0, false, false);
        acc1 = __builtin_amdgcn_wmma_f32_16x16x32_f16(false, a_hi[1], false, bhi1,
                                                      (short)0, acc1, false, false);
        acc0 = __builtin_amdgcn_wmma_f32_16x16x32_f16(false, a_lo[0], false, bhi0,
                                                      (short)0, acc0, false, false);
        acc1 = __builtin_amdgcn_wmma_f32_16x16x32_f16(false, a_lo[1], false, bhi1,
                                                      (short)0, acc1, false, false);
        acc0 = __builtin_amdgcn_wmma_f32_16x16x32_f16(false, a_hi[0], false, blo0,
                                                      (short)0, acc0, false, false);
        acc1 = __builtin_amdgcn_wmma_f32_16x16x32_f16(false, a_hi[1], false, blo1,
                                                      (short)0, acc1, false, false);

        const float c2n = c2s[n];
        #pragma unroll
        for (int r = 0; r < 8; ++r) {                   // acc[r]: row m0 + r + 8*(lane/16)
            float d = fmaf(-2.0f, acc0[r] + acc1[r], c2n);
            if (d < minv[r]) { minv[r] = d; mini[r] = n; }
        }
    }

    // ---- Stage 4: argmin reduction across the 16 lanes of each half-wave
    #pragma unroll
    for (int r = 0; r < 8; ++r) {
        #pragma unroll
        for (int off = 8; off >= 1; off >>= 1) {
            float ov = __shfl_xor(minv[r], off, 32);
            int   oi = __shfl_xor(mini[r], off, 32);
            if (ov < minv[r] || (ov == minv[r] && oi < mini[r])) {
                minv[r] = ov; mini[r] = oi;
            }
        }
    }
    // lanes 0-15 hold codes for rows m0+r; lanes 16-31 for rows m0+8+r.
    // Bounce through LDS so every lane can fetch its own row's code
    // (same-wave DS ops are in-order, so no barrier needed).
    if (lane == 0) {
        #pragma unroll
        for (int r = 0; r < 8; ++r) wave_codes[wave][r] = mini[r];
    }
    if (lane == 16) {
        #pragma unroll
        for (int r = 0; r < 8; ++r) wave_codes[wave][8 + r] = mini[r];
    }
    const int mycode = wave_codes[wave][lane & 15];     // code for row m
    if (lane < 16) out_codes[m0 + lane] = (float)wave_codes[wave][lane];

    // ---- Stage 5: gather exact fp32 codebook row, write quantized in A-layout,
    //      accumulate SSE from hi+lo reconstructed latents (no latents re-read).
    const float* qrow = codebook + (size_t)mycode * CODE_DIM;
    float*       orow = out_q    + (size_t)m      * CODE_DIM;
    float sse = 0.f;
    #pragma unroll
    for (int s = 0; s < 2; ++s) {
        #pragma unroll
        for (int h = 0; h < 2; ++h) {
            const int c0 = 32 * s + 8 * h + kb;         // 8-float contiguous chunk
            float4 qa = *(const float4*)(qrow + c0);
            float4 qb = *(const float4*)(qrow + c0 + 4);
            *(float4*)(orow + c0)     = qa;
            *(float4*)(orow + c0 + 4) = qb;
            const float qs[8] = {qa.x, qa.y, qa.z, qa.w, qb.x, qb.y, qb.z, qb.w};
            #pragma unroll
            for (int j = 0; j < 8; ++j) {
                float x = (float)a_hi[s][8*h + j] + (float)a_lo[s][8*h + j];
                float d = x - qs[j];
                sse = fmaf(d, d, sse);
            }
        }
    }
    #pragma unroll
    for (int off = 16; off >= 1; off >>= 1) sse += __shfl_xor(sse, off, 32);
    if (lane == 0) wave_partial[wave] = sse;
    __syncthreads();
    if (tid == 0) {
        float s = 0.f;
        #pragma unroll
        for (int i = 0; i < 8; ++i) s += wave_partial[i];
        block_partials[blockIdx.x] = s;
    }
}

// Deterministic fixed-order reduction of block partials -> vq_loss scalar.
__global__ __launch_bounds__(256) void vq_finalize_kernel(
    const float* __restrict__ partials, int n, float* __restrict__ loss_out,
    float scale)
{
    __shared__ float s[256];
    float acc = 0.f;
    for (int i = threadIdx.x; i < n; i += 256) acc += partials[i];
    s[threadIdx.x] = acc;
    __syncthreads();
    for (int stride = 128; stride > 0; stride >>= 1) {
        if (threadIdx.x < stride) s[threadIdx.x] += s[threadIdx.x + stride];
        __syncthreads();
    }
    if (threadIdx.x == 0) *loss_out = s[0] * scale;
}

extern "C" void kernel_launch(void* const* d_in, const int* in_sizes, int n_in,
                              void* d_out, int out_size, void* d_ws, size_t ws_size,
                              hipStream_t stream) {
    const float* latents  = (const float*)d_in[0];
    const float* codebook = (const float*)d_in[1];
    const int rows = in_sizes[0] / CODE_DIM;            // 131072

    float* out_q     = (float*)d_out;                    // rows*64
    float* out_codes = out_q + (size_t)rows * CODE_DIM;  // rows
    float* out_loss  = out_codes + rows;                 // 1
    float* partials  = (float*)d_ws;

    const int blocks = rows / (16 * 8);                  // 1024
    vq_wmma_kernel<<<blocks, 256, 0, stream>>>(latents, codebook, out_q, out_codes,
                                               partials);
    const float scale = 1.25f / (float)((size_t)rows * CODE_DIM);
    vq_finalize_kernel<<<1, 256, 0, stream>>>(partials, blocks, out_loss, scale);
}
